// Th1nker_29901562315181
// MI455X (gfx1250) — matrix operand order repaired and checked by reading, not compiled, not run on hardware
//
#include <hip/hip_runtime.h>
#include <hip/hip_bf16.h>
#include <math.h>

// ---------------------------------------------------------------------------
// MI455X (gfx1250) latent-transformer forward pass.
// - All matmul FLOPs on V_WMMA_F32_16X16X32_F16 (f16 in, f32 accumulate).
// - GEMM: 128x64 tiles, register double-buffered global->LDS pipeline,
//   clamp-not-predicate ragged M, f16 output mode for attention operands.
// - Attention: flash-style, one block per (b,head,q-half); 8 waves share
//   K/V chunks staged to LDS with GLOBAL_LOAD_ASYNC_TO_LDS_B128 (ASYNCcnt),
//   double-buffered; ~16x less K/V traffic than per-q-tile streaming.
// Workspace requirement: ~108 MiB.
// ---------------------------------------------------------------------------

typedef _Float16 h16;
typedef __attribute__((ext_vector_type(16))) _Float16 v16h;
typedef __attribute__((ext_vector_type(8)))  float    v8f;

#define WMMA16(a, b, c) \
  __builtin_amdgcn_wmma_f32_16x16x32_f16(false, (a), false, (b), (short)0, (c), false, false)

#define MODE_F32  0
#define MODE_GELU 1
#define MODE_F16  2

// A-fragment (16x32 f16) K index for vgpr-pair vv, lane-half hi (contiguous: k = 16*hi + 2*vv)
__device__ __forceinline__ int afrag_k(int vv, int hi) {
  return 2 * vv + 16 * hi;
}

__device__ __forceinline__ float block_reduce_sum(float v, float* red) {
  int tid = threadIdx.x;
  red[tid] = v;
  __syncthreads();
  for (int s = 128; s > 0; s >>= 1) {
    if (tid < s) red[tid] += red[tid + s];
    __syncthreads();
  }
  float r = red[0];
  __syncthreads();
  return r;
}

// ---------------------------------------------------------------------------
// GEMM: C = op(A[M,K] @ W[K,N] + bias).  N mult of 64, K mult of 32, M ragged.
// 128 threads = 4 waves; 128x64 tile; K-step 32; reg double-buffering.
// ---------------------------------------------------------------------------
__global__ __launch_bounds__(128)
void gemm_wmma(const float* __restrict__ A, const float* __restrict__ W,
               const float* __restrict__ bias, void* __restrict__ C,
               int M, int N, int K, int mode) {
  __shared__ h16 As[128][34];  // [m][k]
  __shared__ h16 Bs[64][34];   // [n][k]  (transposed: K-pairs contiguous)

  const int tid  = threadIdx.x;
  const int lane = tid & 31;
  const int wave = tid >> 5;
  const int l15  = lane & 15;
  const int hi   = lane >> 4;
  const int n0   = blockIdx.x * 64;
  const int m0   = blockIdx.y * 128;

  const int ar[8] = {
      (tid + 0 * 128) >> 3, (tid + 1 * 128) >> 3, (tid + 2 * 128) >> 3,
      (tid + 3 * 128) >> 3, (tid + 4 * 128) >> 3, (tid + 5 * 128) >> 3,
      (tid + 6 * 128) >> 3, (tid + 7 * 128) >> 3};
  const int ac4 = tid & 7;
  const int bk[4] = {
      (tid + 0 * 128) >> 4, (tid + 1 * 128) >> 4, (tid + 2 * 128) >> 4,
      (tid + 3 * 128) >> 4};
  const int bn4 = tid & 15;

  float4 abuf[8], bbuf[4];

  auto load_tiles = [&](int k0) {
#pragma unroll
    for (int i = 0; i < 8; ++i) {
      int gm = m0 + ar[i];
      gm = (gm < M) ? gm : (M - 1);   // clamped rows never stored
      abuf[i] = *(const float4*)&A[(size_t)gm * K + k0 + ac4 * 4];
    }
#pragma unroll
    for (int i = 0; i < 4; ++i)
      bbuf[i] = *(const float4*)&W[(size_t)(k0 + bk[i]) * N + n0 + bn4 * 4];
  };
  auto store_tiles = [&]() {
#pragma unroll
    for (int i = 0; i < 8; ++i) {
      As[ar[i]][ac4 * 4 + 0] = (h16)abuf[i].x;
      As[ar[i]][ac4 * 4 + 1] = (h16)abuf[i].y;
      As[ar[i]][ac4 * 4 + 2] = (h16)abuf[i].z;
      As[ar[i]][ac4 * 4 + 3] = (h16)abuf[i].w;
    }
#pragma unroll
    for (int i = 0; i < 4; ++i) {
      Bs[bn4 * 4 + 0][bk[i]] = (h16)bbuf[i].x;
      Bs[bn4 * 4 + 1][bk[i]] = (h16)bbuf[i].y;
      Bs[bn4 * 4 + 2][bk[i]] = (h16)bbuf[i].z;
      Bs[bn4 * 4 + 3][bk[i]] = (h16)bbuf[i].w;
    }
  };

  v8f acc[2][4];
#pragma unroll
  for (int u = 0; u < 2; ++u)
#pragma unroll
    for (int s = 0; s < 4; ++s) acc[u][s] = {};

  load_tiles(0);
  store_tiles();
  __syncthreads();

  for (int k0 = 0; k0 < K; k0 += 32) {
    const bool has_next = (k0 + 32 < K);
    if (has_next) load_tiles(k0 + 32);  // b128s in flight under the WMMAs

    v16h afr[2];
#pragma unroll
    for (int u = 0; u < 2; ++u) {
      const int mrow = wave * 32 + u * 16 + l15;
#pragma unroll
      for (int vv = 0; vv < 8; ++vv) {
        int k = afrag_k(vv, hi);
        afr[u][2 * vv]     = As[mrow][k];
        afr[u][2 * vv + 1] = As[mrow][k + 1];
      }
    }
#pragma unroll
    for (int s = 0; s < 4; ++s) {
      v16h bfr;
      int n = s * 16 + l15;
#pragma unroll
      for (int vv = 0; vv < 8; ++vv) {
        int k = 2 * vv + 16 * hi;
        bfr[2 * vv]     = Bs[n][k];
        bfr[2 * vv + 1] = Bs[n][k + 1];
      }
      acc[0][s] = WMMA16(afr[0], bfr, acc[0][s]);
      acc[1][s] = WMMA16(afr[1], bfr, acc[1][s]);
    }
    __syncthreads();
    if (has_next) store_tiles();
    __syncthreads();
  }

#pragma unroll
  for (int u = 0; u < 2; ++u) {
#pragma unroll
    for (int s = 0; s < 4; ++s) {
      int col = n0 + s * 16 + l15;
      float bcol = bias[col];
#pragma unroll
      for (int vv = 0; vv < 8; ++vv) {
        int row = m0 + wave * 32 + u * 16 + vv + 8 * hi;
        if (row < M) {
          float x = acc[u][s][vv] + bcol;
          if (mode == MODE_GELU) {
            float t = 0.7978845608028654f * (x + 0.044715f * x * x * x);
            x = 0.5f * x * (1.0f + tanhf(t));
          }
          if (mode == MODE_F16)
            ((h16*)C)[(size_t)row * N + col] = (h16)x;
          else
            ((float*)C)[(size_t)row * N + col] = x;
        }
      }
    }
  }
}

// ---------------------------------------------------------------------------
// Flash attention.  Grid: 128 blocks = (b, head, q-half); 256 thr = 8 waves,
// each wave owns 2 query tiles of 16.  Q=512, K=2816, HS=64, all f16 inputs.
// Keys 0..2047 KVin | 2048..2303 KVmem | 2304..2815 QKV(k_new/v_new).
// Mask: visible iff kk < 2561 || r + kk < 2816.  1/sqrt(64) applied to S.
// K/V chunks (32 keys) staged once per block via async loads, double-buffered.
// ---------------------------------------------------------------------------
__global__ __launch_bounds__(256)
void attn_wmma(const h16* __restrict__ QKV, const h16* __restrict__ KVin,
               const h16* __restrict__ KVmem, float* __restrict__ Out) {
  __shared__ __align__(16) h16 Qs[256][64];     // 32 KB
  __shared__ __align__(16) h16 Ks[2][32][64];   // 8 KB (double buffer)
  __shared__ __align__(16) h16 Vs[2][32][64];   // 8 KB
  __shared__ float Sb[8][16][32];               // 16 KB, per-wave scratch
  __shared__ h16   Pb[8][16][32];               // 8 KB
  __shared__ float corrb[8][16];
  __shared__ float lb[8][16];

  const int tid  = threadIdx.x;
  const int lane = tid & 31;
  const int w    = tid >> 5;
  const int l15  = lane & 15;
  const int hi   = lane >> 4;
  const int qhalf = blockIdx.x & 1;
  const int h     = (blockIdx.x >> 1) & 15;
  const int b     = blockIdx.x >> 5;

  // stage this block's 256 queries (raw f16 copy)
#pragma unroll
  for (int i = 0; i < 8; ++i) {
    int e = tid + i * 256;
    int q = e >> 3, j = e & 7;
    const uint4* src =
        (const uint4*)(QKV + (size_t)(b * 512 + qhalf * 256 + q) * 3072 + h * 64);
    *(uint4*)&Qs[q][j * 8] = src[j];
  }

  // wave w async-copies its 512B slice of the 4KB K tile (and V tile)
  auto issue_chunk = [&](int c, int d) {
    const h16 *kb, *vb;
    int stride;
    if (c < 64) {
      kb = KVin + (size_t)(b * 2048 + c * 32) * 2048 + h * 64;
      vb = kb + 1024; stride = 2048;
    } else if (c < 72) {
      kb = KVmem + (size_t)(b * 256 + (c - 64) * 32) * 2048 + h * 64;
      vb = kb + 1024; stride = 2048;
    } else {
      kb = QKV + (size_t)(b * 512 + (c - 72) * 32) * 3072 + 1024 + h * 64;
      vb = kb + 1024; stride = 3072;
    }
    int key = w * 4 + (lane >> 3);
    int off = (lane & 7) * 8;  // halves
    const h16* gk = kb + (size_t)key * stride + off;
    const h16* gv = vb + (size_t)key * stride + off;
    unsigned lk = (unsigned)(unsigned long long)&Ks[d][key][off];
    unsigned lv = (unsigned)(unsigned long long)&Vs[d][key][off];
    asm volatile("global_load_async_to_lds_b128 %0, %1, off"
                 :: "v"(lk), "v"(gk) : "memory");
    asm volatile("global_load_async_to_lds_b128 %0, %1, off"
                 :: "v"(lv), "v"(gv) : "memory");
  };

  issue_chunk(0, 0);
  asm volatile("s_wait_asynccnt 0x0" ::: "memory");
  __syncthreads();

  float mrow[2] = {-1e30f, -1e30f};
  float lrow[2] = {0.0f, 0.0f};
  v8f oacc[2][4];
#pragma unroll
  for (int t = 0; t < 2; ++t)
#pragma unroll
    for (int s = 0; s < 4; ++s) oacc[t][s] = {};

  for (int c = 0; c < 88; ++c) {
    const int d = c & 1;
    if (c + 1 < 88) issue_chunk(c + 1, d ^ 1);  // prefetch next chunk (async)

#pragma unroll
    for (int t = 0; t < 2; ++t) {
      const int tl = w * 2 + t;  // tile index within block (0..15)

      // S = Q @ K^T  (two 16x16 key tiles, k-dim 64 = 2 WMMA each)
#pragma unroll
      for (int st = 0; st < 2; ++st) {
        v8f sacc = {};
#pragma unroll
        for (int ks = 0; ks < 2; ++ks) {
          v16h qa, kf;
#pragma unroll
          for (int vv = 0; vv < 8; ++vv) {
            int ka = afrag_k(vv, hi) + 32 * ks;
            qa[2 * vv]     = Qs[tl * 16 + l15][ka];
            qa[2 * vv + 1] = Qs[tl * 16 + l15][ka + 1];
            int kbk = 2 * vv + 16 * hi + 32 * ks;
            kf[2 * vv]     = Ks[d][st * 16 + l15][kbk];
            kf[2 * vv + 1] = Ks[d][st * 16 + l15][kbk + 1];
          }
          sacc = WMMA16(qa, kf, sacc);
        }
#pragma unroll
        for (int vv = 0; vv < 8; ++vv)
          Sb[w][vv + 8 * hi][st * 16 + l15] = sacc[vv] * 0.125f;
      }

      // online softmax: lanes 0..15 own one query row each (per-wave LDS,
      // DS ops are in-order within a wave -> no barrier needed)
      if (lane < 16) {
        int r = qhalf * 256 + tl * 16 + lane;
        float mx = mrow[t];
        float sv[32];
#pragma unroll 8
        for (int j = 0; j < 32; ++j) {
          int kk = c * 32 + j;
          float s = Sb[w][lane][j];
          bool ok = (kk < 2561) || (r + kk < 2816);
          s = ok ? s : -1e30f;
          sv[j] = s;
          mx = fmaxf(mx, s);
        }
        float corr = __expf(mrow[t] - mx);
        float sum = 0.0f;
#pragma unroll 8
        for (int j = 0; j < 32; ++j) {
          float p = __expf(sv[j] - mx);
          sum += p;
          Pb[w][lane][j] = (h16)p;
        }
        lrow[t] = lrow[t] * corr + sum;
        mrow[t] = mx;
        corrb[w][lane] = corr;
      }

      // rescale O, then O += P @ V
#pragma unroll
      for (int s = 0; s < 4; ++s)
#pragma unroll
        for (int vv = 0; vv < 8; ++vv) oacc[t][s][vv] *= corrb[w][vv + 8 * hi];

      v16h pa;
#pragma unroll
      for (int vv = 0; vv < 8; ++vv) {
        int ka = afrag_k(vv, hi);
        pa[2 * vv]     = Pb[w][l15][ka];
        pa[2 * vv + 1] = Pb[w][l15][ka + 1];
      }
#pragma unroll
      for (int s = 0; s < 4; ++s) {
        v16h vf;
#pragma unroll
        for (int vv = 0; vv < 8; ++vv) {
          int kv = 2 * vv + 16 * hi;  // key index; Vs is [key][hs]
          vf[2 * vv]     = Vs[d][kv][s * 16 + l15];
          vf[2 * vv + 1] = Vs[d][kv + 1][s * 16 + l15];
        }
        oacc[t][s] = WMMA16(pa, vf, oacc[t][s]);
      }
    }

    asm volatile("s_wait_asynccnt 0x0" ::: "memory");  // next chunk landed
    __syncthreads();
  }

#pragma unroll
  for (int t = 0; t < 2; ++t) {
    if (lane < 16) lb[w][lane] = lrow[t];
    const int tl = w * 2 + t;
#pragma unroll
    for (int s = 0; s < 4; ++s)
#pragma unroll
      for (int vv = 0; vv < 8; ++vv) {
        int row = vv + 8 * hi;
        int col = s * 16 + l15;
        Out[(size_t)(b * 512 + qhalf * 256 + tl * 16 + row) * 1024 + h * 64 + col] =
            oacc[t][s][vv] / lb[w][row];
      }
  }
}

// ---------------------------------------------------------------------------
// Elementwise / small kernels
// ---------------------------------------------------------------------------
__global__ __launch_bounds__(256)
void build_x(const float* __restrict__ emb_in, const int* __restrict__ toks,
             float* __restrict__ X) {
  int row = blockIdx.x;             // b*2048 + t
  int t = row & 2047;
  const float* e = emb_in + (size_t)toks[row] * 1024;
  float* dst = X + (size_t)row * 1024;
  float pos = (float)t;
  const float lc = 8.317766166719343f / 1024.0f;  // log(4096)/H
  for (int col = threadIdx.x; col < 1024; col += 256) {
    float d = expf(-(float)(col & ~1) * lc);
    float pe = (col & 1) ? cosf(pos * d) : sinf(pos * d);
    dst[col] = e[col] + pe;
  }
}

__global__ __launch_bounds__(256)
void build_latent(const float* __restrict__ noise, const float* __restrict__ emb_lat,
                  float* __restrict__ latent) {
  size_t i = (size_t)blockIdx.x * 256 + threadIdx.x;
  latent[i] = noise[i] + emb_lat[i & 1023];
}

__global__ __launch_bounds__(256)
void build_latcat(const float* __restrict__ latent, const float* __restrict__ emb_query,
                  const float* __restrict__ emb_causal, const int* __restrict__ otoks,
                  const float* __restrict__ g, const float* __restrict__ bb,
                  float* __restrict__ latcat) {
  __shared__ float red[256];
  int row = blockIdx.x;
  int b = row >> 9, rl = row & 511;
  float* dst = latcat + (size_t)row * 1024;
  if (rl < 256) {  // layer-normed latent
    const float* src = latent + (size_t)(b * 256 + rl) * 1024;
    float s = 0.0f;
    for (int i = threadIdx.x; i < 1024; i += 256) s += src[i];
    float mu = block_reduce_sum(s, red) * (1.0f / 1024.0f);
    float vs = 0.0f;
    for (int i = threadIdx.x; i < 1024; i += 256) { float d = src[i] - mu; vs += d * d; }
    float var = block_reduce_sum(vs, red) * (1.0f / 1024.0f);
    float rstd = rsqrtf(var + 1e-5f);
    for (int i = threadIdx.x; i < 1024; i += 256)
      dst[i] = (src[i] - mu) * rstd * g[i] + bb[i];
  } else {
    const float* src;
    if (rl == 256)      src = emb_query;
    else if (rl == 257) src = emb_query + 1024;
    else                src = emb_causal + (size_t)otoks[b * 255 + (rl - 258)] * 1024;
    for (int i = threadIdx.x; i < 1024; i += 256) dst[i] = src[i];
  }
}

__global__ __launch_bounds__(256)
void layernorm(const float* __restrict__ in, float* __restrict__ out,
               const float* __restrict__ g, const float* __restrict__ bb) {
  __shared__ float red[256];
  const float* src = in + (size_t)blockIdx.x * 1024;
  float* dst = out + (size_t)blockIdx.x * 1024;
  float s = 0.0f;
  for (int i = threadIdx.x; i < 1024; i += 256) s += src[i];
  float mu = block_reduce_sum(s, red) * (1.0f / 1024.0f);
  float vs = 0.0f;
  for (int i = threadIdx.x; i < 1024; i += 256) { float d = src[i] - mu; vs += d * d; }
  float var = block_reduce_sum(vs, red) * (1.0f / 1024.0f);
  float rstd = rsqrtf(var + 1e-5f);
  for (int i = threadIdx.x; i < 1024; i += 256)
    dst[i] = (src[i] - mu) * rstd * g[i] + bb[i];
}

__global__ __launch_bounds__(256)
void add_inplace(float* __restrict__ a, const float* __restrict__ b) {
  size_t i = (size_t)blockIdx.x * 256 + threadIdx.x;
  a[i] += b[i];
}

__global__ __launch_bounds__(256)
void copy_tail(const float* __restrict__ latout, float* __restrict__ tail) {
  int row = blockIdx.x;  // 0..1019
  int b = row / 255, t = row % 255;
  const float* src = latout + (size_t)(b * 512 + 257 + t) * 1024;
  float* dst = tail + (size_t)row * 1024;
  for (int i = threadIdx.x; i < 1024; i += 256) dst[i] = src[i];
}

__global__ __launch_bounds__(256)
void rowdot(const float* __restrict__ X, const float* __restrict__ w,
            const float* __restrict__ bias, float* __restrict__ out, size_t xstride) {
  __shared__ float red[256];
  const float* x = X + (size_t)blockIdx.x * xstride;
  float s = 0.0f;
  for (int i = threadIdx.x; i < 1024; i += 256) s += x[i] * w[i];
  s = block_reduce_sum(s, red);
  if (threadIdx.x == 0) out[blockIdx.x] = s + bias[0];
}

// ---------------------------------------------------------------------------
extern "C" void kernel_launch(void* const* d_in, const int* in_sizes, int n_in,
                              void* d_out, int out_size, void* d_ws, size_t ws_size,
                              hipStream_t stream) {
  const float* noise      = (const float*)d_in[0];
  const float* emb_in     = (const float*)d_in[1];
  const float* emb_causal = (const float*)d_in[2];
  const float* emb_query  = (const float*)d_in[3];
  const float* emb_latent = (const float*)d_in[4];
  const float* W_kv_in    = (const float*)d_in[5];
  const float* b_kv_in    = (const float*)d_in[6];
  const float* W_kv_mem   = (const float*)d_in[7];
  const float* b_kv_mem   = (const float*)d_in[8];
  const float* W_sc       = (const float*)d_in[9];
  const float* b_sc       = (const float*)d_in[10];
  const float* g1         = (const float*)d_in[11];
  const float* b1         = (const float*)d_in[12];
  const float* g2         = (const float*)d_in[13];
  const float* b2         = (const float*)d_in[14];
  const float* W_fc       = (const float*)d_in[15];
  const float* b_fc       = (const float*)d_in[16];
  const float* W_proj     = (const float*)d_in[17];
  const float* b_proj     = (const float*)d_in[18];
  const float* W_probe    = (const float*)d_in[19];
  const float* b_probe    = (const float*)d_in[20];
  const float* W_oprobe   = (const float*)d_in[21];
  const float* b_oprobe   = (const float*)d_in[22];
  const float* W_out      = (const float*)d_in[23];
  const float* b_out      = (const float*)d_in[24];
  const int*   in_toks    = (const int*)d_in[25];
  const int*   out_toks   = (const int*)d_in[26];
  float* out = (float*)d_out;

  // workspace layout (bytes), total 113,246,208 (~108 MiB)
  char* p = (char*)d_ws;
  float* X       = (float*)p;  p += (size_t)8388608 * 4;   // X; later H1/H2
  h16*   KVin16  = (h16*)p;    p += (size_t)16777216 * 2;  // k_in|v_in (f16)
  h16*   KVmem16 = (h16*)p;    p += (size_t)2097152 * 2;   // k_mem|v_mem (f16)
  float* LatBuf  = (float*)p;  p += (size_t)1048576 * 4;   // latent; later Tail
  float* LatCat  = (float*)p;  p += (size_t)2097152 * 4;   // lat_cat -> lat_out
  h16*   QKV16   = (h16*)p;
  float* ZLN     = (float*)QKV16;                          // overlays after attn
  p += (size_t)6291456 * 2;
  float* AttnB   = (float*)p;  p += (size_t)2097152 * 4;   // attn out; later Z
  float* M2      = (float*)p;  p += (size_t)2097152 * 4;

  float* H    = X;
  float* Z    = AttnB;
  float* Tail = LatBuf;

  float* probe_out = out;
  float* logits    = out + 4;
  float* oprobe    = out + 4 + 32640000;

  build_x<<<8192, 256, 0, stream>>>(emb_in, in_toks, X);
  build_latent<<<4096, 256, 0, stream>>>(noise, emb_latent, LatBuf);
  gemm_wmma<<<dim3(32, 64), 128, 0, stream>>>(X, W_kv_in, b_kv_in, KVin16, 8192, 2048, 1024, MODE_F16);
  gemm_wmma<<<dim3(32, 8), 128, 0, stream>>>(LatBuf, W_kv_mem, b_kv_mem, KVmem16, 1024, 2048, 1024, MODE_F16);
  build_latcat<<<2048, 256, 0, stream>>>(LatBuf, emb_query, emb_causal, out_toks, g1, b1, LatCat);
  gemm_wmma<<<dim3(48, 16), 128, 0, stream>>>(LatCat, W_sc, b_sc, QKV16, 2048, 3072, 1024, MODE_F16);
  attn_wmma<<<128, 256, 0, stream>>>(QKV16, KVin16, KVmem16, AttnB);
  gemm_wmma<<<dim3(64, 16), 128, 0, stream>>>(AttnB, W_fc, b_fc, H, 2048, 4096, 1024, MODE_GELU);
  gemm_wmma<<<dim3(16, 16), 128, 0, stream>>>(H, W_proj, b_proj, Z, 2048, 1024, 4096, MODE_F32);
  layernorm<<<2048, 256, 0, stream>>>(Z, ZLN, g2, b2);
  gemm_wmma<<<dim3(64, 16), 128, 0, stream>>>(ZLN, W_fc, b_fc, H, 2048, 4096, 1024, MODE_GELU);
  gemm_wmma<<<dim3(16, 16), 128, 0, stream>>>(H, W_proj, b_proj, M2, 2048, 1024, 4096, MODE_F32);
  add_inplace<<<8192, 256, 0, stream>>>(LatCat, M2);
  copy_tail<<<1020, 256, 0, stream>>>(LatCat, Tail);
  rowdot<<<4, 256, 0, stream>>>(LatCat + (size_t)256 * 1024, W_probe, b_probe, probe_out,
                                (size_t)512 * 1024);
  gemm_wmma<<<dim3(500, 8), 128, 0, stream>>>(Tail, W_out, b_out, logits, 1020, 32000, 1024, MODE_F32);
  rowdot<<<1020, 256, 0, stream>>>(Tail, W_oprobe, b_oprobe, oprobe, (size_t)1024);
}
